// SelfAttention_71897752535321
// MI455X (gfx1250) — compile-verified, matrix operand order checked by reading, not run on hardware
//
#include <hip/hip_runtime.h>

// ---------------- types ----------------
typedef __bf16 v16bf __attribute__((ext_vector_type(16)));
typedef __bf16 v8bf  __attribute__((ext_vector_type(8)));
typedef __bf16 v4bf  __attribute__((ext_vector_type(4)));
typedef float  v8f   __attribute__((ext_vector_type(8)));
typedef float  v4f   __attribute__((ext_vector_type(4)));
typedef unsigned int u32x4 __attribute__((ext_vector_type(4)));
typedef int i32x8 __attribute__((ext_vector_type(8)));
typedef int i32x4 __attribute__((ext_vector_type(4)));

#define BQ  2
#define SQ  2048
#define DQ  1024
#define HQ  16
#define HDQ 64
#define MQ  (BQ * SQ)        // 4096
#define KQ  DQ               // 1024

// ---------------- helpers ----------------
__device__ __forceinline__ v8f wmma_bf16(v16bf a, v16bf b, v8f c) {
    return __builtin_amdgcn_wmma_f32_16x16x32_bf16(false, a, false, b, (short)0, c, false, false);
}

__device__ __forceinline__ v8f zero_v8f() {
    v8f z;
#pragma unroll
    for (int e = 0; e < 8; ++e) z[e] = 0.0f;
    return z;
}

// Build a 16x32 A/B fragment from two contiguous 8-half (16B) loads.
// element i <-> K = koff + (i<8?0:16) + (lane/16)*8 + i%8
__device__ __forceinline__ v16bf frag_from_row(const __bf16* rp, int koff, int half) {
    v8bf lo = *(const v8bf*)(rp + koff + half * 8);
    v8bf hi = *(const v8bf*)(rp + koff + 16 + half * 8);
    return __builtin_shufflevector(lo, hi, 0,1,2,3,4,5,6,7,8,9,10,11,12,13,14,15);
}

// A/B fragment from a [rows][64] row-major global bf16 tensor.
__device__ __forceinline__ v16bf load_rowfrag64(const __bf16* base, int row0, int koff) {
    const int lane = threadIdx.x & 31;
    const int l16 = lane & 15, half = lane >> 4;
    return frag_from_row(base + (size_t)(row0 + l16) * HDQ, koff, half);
}

// TDM: async-load a 2D bf16 tile [tile_rows x tile_cols] (row stride = tensor_stride elems)
// into LDS with 16B padding after every 64B row chunk -> LDS row stride 80B (LDA=40 halves).
// Issue from ONE wave (wave-uniform args); tracked by TENSORcnt.
__device__ __forceinline__ void tdm_load_tile_2d(unsigned lds_byte_off, const void* gaddr,
                                                 unsigned tile_rows, unsigned tile_cols,
                                                 unsigned tensor_stride) {
    unsigned long long ga = (unsigned long long)(size_t)gaddr;
    u32x4 g0;
    g0.x = 1u;                                           // count=1, is_restore=0, gather off
    g0.y = lds_byte_off;                                 // D#.lds_addr (bytes)
    g0.z = (unsigned)(ga & 0xffffffffu);                 // global_addr[31:0]
    g0.w = (unsigned)((ga >> 32) & 0x1ffffffu)           // global_addr[56:32]
         | (2u << 30);                                   // type = 2 ("image")
    i32x8 g1;
    g1[0] = (1 << 16)                                    // data_size = 2B
          | (1 << 20)                                    // pad_enable
          | (3 << 22)                                    // pad_interval: 16 DWORDs = 64B
          | (3 << 25);                                   // pad_amount:   4 DWORDs = 16B
    g1[1] = (int)((tensor_stride & 0xffffu) << 16);      // tensor_dim0 low16 (= row length)
    g1[2] = (int)(((tensor_stride >> 16) & 0xffffu)      // tensor_dim0 high16
          | (0xffffu << 16));                            // tensor_dim1 low16 (large)
    g1[3] = (int)((tile_cols & 0xffffu) << 16);          // tensor_dim1 hi16=0 | tile_dim0
    g1[4] = (int)(tile_rows & 0xffffu);                  // tile_dim1 | tile_dim2=0
    g1[5] = (int)tensor_stride;                          // tensor_dim0_stride low32
    g1[6] = 0;                                           // stride hi16 | tensor_dim1_stride lo
    g1[7] = 0;
    i32x4 z4 = {0, 0, 0, 0};
#if __clang_major__ >= 23
    i32x8 z8 = {0, 0, 0, 0, 0, 0, 0, 0};
    __builtin_amdgcn_tensor_load_to_lds(g0, g1, z4, z4, z8, 0);
#else
    __builtin_amdgcn_tensor_load_to_lds(g0, g1, z4, z4, 0);
#endif
}

// ---------------- f32 -> bf16 conversion (vectorized x4) ----------------
__global__ void f32_to_bf16_kernel(const float* __restrict__ src,
                                   unsigned short* __restrict__ dst_, int n4) {
    __bf16* dst = (__bf16*)dst_;
    int i = blockIdx.x * blockDim.x + threadIdx.x;
    if (i < n4) {
        v4f v = *(const v4f*)(src + (size_t)i * 4);
        v4bf o;
#pragma unroll
        for (int u = 0; u < 4; ++u) o[u] = (__bf16)v[u];
        *(v4bf*)(dst + (size_t)i * 4) = o;
    }
}

// ---------------- f32 [K][N] -> bf16 [N][K] tiled transpose-convert ----------------
__global__ __launch_bounds__(256)
void transpose_convert_kernel(const float* __restrict__ w, unsigned short* __restrict__ wt_,
                              int K, int N) {
    __bf16* wt = (__bf16*)wt_;
    __shared__ float tile[32][33];
    const int n0 = blockIdx.x * 32, k0 = blockIdx.y * 32;
    const int tx = threadIdx.x & 31, ty = threadIdx.x >> 5;   // 32 x 8
#pragma unroll
    for (int i = 0; i < 4; ++i)
        tile[ty + i * 8][tx] = w[(size_t)(k0 + ty + i * 8) * N + n0 + tx];
    __syncthreads();
#pragma unroll
    for (int i = 0; i < 4; ++i)
        wt[(size_t)(n0 + ty + i * 8) * K + k0 + tx] = (__bf16)tile[tx][ty + i * 8];
}

// ---------------- tiled WMMA GEMM, TDM-staged double-buffered LDS ----------------
// C[4096xN] = A[4096x1024] * Wt[Nx1024]^T + bias  (Wt is pre-transposed, row-major [N][K])
// mode 0: qkv epilogue (scatter to Q/K/V bf16 + fp32 present);  mode 1: fp32 store
__global__ __launch_bounds__(256)
void wmma_gemm_kernel(const unsigned short* __restrict__ A_, const unsigned short* __restrict__ Wt_,
                      const float* __restrict__ bias, int N, int mode,
                      float* __restrict__ outp,
                      unsigned short* __restrict__ qbf_, unsigned short* __restrict__ kbf_,
                      unsigned short* __restrict__ vbf_, float* __restrict__ present) {
    const __bf16* A  = (const __bf16*)A_;
    const __bf16* Wt = (const __bf16*)Wt_;
    __bf16* qbf = (__bf16*)qbf_;
    __bf16* kbf = (__bf16*)kbf_;
    __bf16* vbf = (__bf16*)vbf_;

    constexpr int LDA = 40;                       // 80B LDS row stride (TDM pad produces this)
    constexpr int NK  = KQ / 32;
    __shared__ __align__(128) __bf16 As[2][128 * LDA];
    __shared__ __align__(128) __bf16 Bs[2][128 * LDA];   // rows are N (Wt rows)

    const int tid  = threadIdx.x;
    const int lane = tid & 31;
    const int wave = tid >> 5;                    // 8 waves (wave32)
    const int wm = wave & 1, wn = wave >> 1;      // 2x4 wave grid -> 64x32 per wave
    const int l16 = lane & 15, half = lane >> 4;
    const int bm = blockIdx.y * 128;
    const int bn = blockIdx.x * 128;

    auto issue_tile = [&](int kt, int buf) {
        const int k0 = kt * 32;
        tdm_load_tile_2d((unsigned)(size_t)&As[buf][0], A  + (size_t)bm * KQ + k0, 128, 32, KQ);
        tdm_load_tile_2d((unsigned)(size_t)&Bs[buf][0], Wt + (size_t)bn * KQ + k0, 128, 32, KQ);
    };

    v8f acc[4][2];
#pragma unroll
    for (int i = 0; i < 4; ++i)
#pragma unroll
        for (int j = 0; j < 2; ++j) acc[i][j] = zero_v8f();

    if (wave == 0) issue_tile(0, 0);

    for (int kt = 0; kt < NK; ++kt) {
        if (wave == 0) {
            if (kt + 1 < NK) {
                issue_tile(kt + 1, (kt + 1) & 1);           // DMA next tile into other buffer
                __builtin_amdgcn_s_wait_tensorcnt(2);       // tile kt's 2 TDM ops complete
            } else {
                __builtin_amdgcn_s_wait_tensorcnt(0);
            }
        }
        __syncthreads();                                    // publish tile kt
        const int buf = kt & 1;

        v16bf af[4];
#pragma unroll
        for (int mt = 0; mt < 4; ++mt)
            af[mt] = frag_from_row(&As[buf][(wm * 64 + mt * 16 + l16) * LDA], 0, half);
        v16bf bfr[2];
#pragma unroll
        for (int nt = 0; nt < 2; ++nt)
            bfr[nt] = frag_from_row(&Bs[buf][(wn * 32 + nt * 16 + l16) * LDA], 0, half);
#pragma unroll
        for (int mt = 0; mt < 4; ++mt)
#pragma unroll
            for (int nt = 0; nt < 2; ++nt)
                acc[mt][nt] = wmma_bf16(af[mt], bfr[nt], acc[mt][nt]);
        __syncthreads();                                    // compute done: next DMA may overwrite
    }

    // epilogue
#pragma unroll
    for (int mt = 0; mt < 4; ++mt)
#pragma unroll
        for (int nt = 0; nt < 2; ++nt)
#pragma unroll
            for (int r = 0; r < 8; ++r) {
                int grow = bm + wm * 64 + mt * 16 + half * 8 + r;   // M index (b*S+s)
                int gcol = bn + wn * 32 + nt * 16 + l16;            // N index
                float v = acc[mt][nt][r] + bias[gcol];
                if (mode == 1) {
                    outp[(size_t)grow * N + gcol] = v;
                } else {
                    int bidx = grow >> 11, s = grow & (SQ - 1);
                    int which = gcol >> 10;
                    int rem = gcol & 1023;
                    int h = rem >> 6, hd = rem & 63;
                    size_t kvidx = (((size_t)(bidx * HQ + h)) * SQ + s) * HDQ + hd;
                    if (which == 0) {
                        qbf[kvidx] = (__bf16)v;
                    } else if (which == 1) {
                        kbf[kvidx] = (__bf16)v;
                        present[kvidx] = v;   // present[b] = K
                    } else {
                        vbf[kvidx] = (__bf16)v;
                        present[(((size_t)((BQ + bidx) * HQ + h)) * SQ + s) * HDQ + hd] = v;  // present[B+b] = V
                    }
                }
            }
}

// ---------------- flash attention: 4 waves x 16 query rows, cooperative K/V staging ----------------
__global__ __launch_bounds__(128)
void flash_attn_kernel(const unsigned short* __restrict__ Q_, const unsigned short* __restrict__ K_,
                       const unsigned short* __restrict__ V_, unsigned short* __restrict__ O_) {
    const __bf16* Qb = (const __bf16*)Q_;
    const __bf16* Kb = (const __bf16*)K_;
    const __bf16* Vb = (const __bf16*)V_;
    __bf16* Ob = (__bf16*)O_;

    constexpr int LDK = 72;   // K tile row stride (144B: 16B aligned, conflict-spread)
    constexpr int LDV = 40;   // V^T tile row stride (80B)
    __shared__ __bf16 Ks[32 * LDK];      // [key][hd]
    __shared__ __bf16 Vt[64 * LDV];      // [hd][key] (transposed)
    __shared__ __bf16 Plds[4][16][40];   // per-wave P bounce (C-layout -> A-layout)

    const int tid = threadIdx.x;
    const int lane = tid & 31, wave = tid >> 5;
    const int l16 = lane & 15, half = lane >> 4;
    const int nqb = SQ / 64;
    const int bh = blockIdx.x / nqb;       // b*H + h
    const int qb = blockIdx.x % nqb;
    const int qbase = qb * 64 + wave * 16;

    const __bf16* qp = Qb + (size_t)bh * SQ * HDQ;
    const __bf16* kp = Kb + (size_t)bh * SQ * HDQ;
    const __bf16* vp = Vb + (size_t)bh * SQ * HDQ;

    const v16bf q0 = load_rowfrag64(qp, qbase, 0);
    const v16bf q1 = load_rowfrag64(qp, qbase, 32);

    float m[8], l[8];
    v8f o[4];
#pragma unroll
    for (int r = 0; r < 8; ++r) { m[r] = -__builtin_inff(); l[r] = 0.0f; }
#pragma unroll
    for (int t = 0; t < 4; ++t) o[t] = zero_v8f();

    const int njb_w   = (qbase + 47) >> 5;        // this wave's causal key-block count
    const int njb_max = (qb * 64 + 95) >> 5;      // workgroup max (uniform trip count)

    for (int j = 0; j < njb_max; ++j) {
        const int kbase = j * 32;
        // cooperative stage: K[32][64] row-major, V^T[64][32]
#pragma unroll
        for (int i = 0; i < 4; ++i) {
            int c = tid + i * 128;                 // 512 chunks of 4 halves
            int key = c >> 4, hd0 = (c & 15) << 2;
            v4bf kv = *(const v4bf*)(kp + (size_t)(kbase + key) * HDQ + hd0);
            v4bf vv = *(const v4bf*)(vp + (size_t)(kbase + key) * HDQ + hd0);
            *(v4bf*)(&Ks[key * LDK + hd0]) = kv;
#pragma unroll
            for (int u = 0; u < 4; ++u) Vt[(hd0 + u) * LDV + key] = vv[u];
        }
        __syncthreads();

        if (j < njb_w) {
            // scores S[16 q][32 keys]: two 16x16 C tiles, k-dim = 64 (two WMMAs each)
            v8f s0 = zero_v8f(), s1 = zero_v8f();
            {
                const __bf16* kr0 = &Ks[(0 + l16) * LDK];
                const __bf16* kr1 = &Ks[(16 + l16) * LDK];
                s0 = wmma_bf16(q0, frag_from_row(kr0, 0, half), s0);
                s0 = wmma_bf16(q1, frag_from_row(kr0, 32, half), s0);
                s1 = wmma_bf16(q0, frag_from_row(kr1, 0, half), s1);
                s1 = wmma_bf16(q1, frag_from_row(kr1, 32, half), s1);
            }

            float alpha[8], p0a[8], p1a[8];
#pragma unroll
            for (int r = 0; r < 8; ++r) {
                const int qi  = qbase + half * 8 + r;
                const int k0i = kbase + l16;
                const int k1i = kbase + 16 + l16;
                float a = s0[r] * 0.125f;  if (k0i > qi) a = -__builtin_inff();
                float b = s1[r] * 0.125f;  if (k1i > qi) b = -__builtin_inff();
                float vmax = fmaxf(a, b);
#pragma unroll
                for (int off = 1; off < 16; off <<= 1) vmax = fmaxf(vmax, __shfl_xor(vmax, off, 32));
                const float mnew = fmaxf(m[r], vmax);
                const float al = __expf(m[r] - mnew);
                const float pa = __expf(a - mnew);
                const float pb = __expf(b - mnew);
                float rs = pa + pb;
#pragma unroll
                for (int off = 1; off < 16; off <<= 1) rs += __shfl_xor(rs, off, 32);
                l[r] = l[r] * al + rs;
                m[r] = mnew;
                alpha[r] = al; p0a[r] = pa; p1a[r] = pb;
            }

            // bounce P through per-wave LDS (same-wave DS ops are in-order)
#pragma unroll
            for (int r = 0; r < 8; ++r) {
                Plds[wave][half * 8 + r][l16]      = (__bf16)p0a[r];
                Plds[wave][half * 8 + r][16 + l16] = (__bf16)p1a[r];
            }
            v16bf pf = frag_from_row(&Plds[wave][l16][0], 0, half);

            // O = O*alpha + P @ V_block  (4 hd tiles, V^T rows are contiguous in k)
#pragma unroll
            for (int t = 0; t < 4; ++t) {
#pragma unroll
                for (int r = 0; r < 8; ++r) o[t][r] *= alpha[r];
                v16bf vf = frag_from_row(&Vt[(t * 16 + l16) * LDV], 0, half);
                o[t] = wmma_bf16(pf, vf, o[t]);
            }
        }
        __syncthreads();
    }

    // normalize and store bf16 attn output in [B*S, D] layout for the proj GEMM
    const int bidx = bh >> 4, h = bh & (HQ - 1);
#pragma unroll
    for (int t = 0; t < 4; ++t)
#pragma unroll
        for (int r = 0; r < 8; ++r) {
            const int s = qbase + half * 8 + r;
            const int col = h * HDQ + t * 16 + l16;
            Ob[((size_t)(bidx * SQ + s)) * DQ + col] = (__bf16)(o[t][r] / l[r]);
        }
}

// ---------------- host ----------------
extern "C" void kernel_launch(void* const* d_in, const int* in_sizes, int n_in,
                              void* d_out, int out_size, void* d_ws, size_t ws_size,
                              hipStream_t stream) {
    const float* x      = (const float*)d_in[0];
    const float* w_qkv  = (const float*)d_in[1];
    const float* b_qkv  = (const float*)d_in[2];
    const float* w_proj = (const float*)d_in[3];
    const float* b_proj = (const float*)d_in[4];

    float* out = (float*)d_out;
    float* present = out + (size_t)BQ * SQ * DQ;

    char* ws = (char*)d_ws;
    unsigned short* xbf     = (unsigned short*)ws; ws += (size_t)MQ * KQ * 2;
    unsigned short* wqkvt   = (unsigned short*)ws; ws += (size_t)KQ * 3 * DQ * 2;  // [3D][K] bf16
    unsigned short* wprojt  = (unsigned short*)ws; ws += (size_t)KQ * DQ * 2;      // [D][K]  bf16
    unsigned short* qbf     = (unsigned short*)ws; ws += (size_t)BQ * HQ * SQ * HDQ * 2;
    unsigned short* kbf     = (unsigned short*)ws; ws += (size_t)BQ * HQ * SQ * HDQ * 2;
    unsigned short* vbf     = (unsigned short*)ws; ws += (size_t)BQ * HQ * SQ * HDQ * 2;
    unsigned short* attnbf  = (unsigned short*)ws;

    const int nx = MQ * KQ;
    f32_to_bf16_kernel<<<(nx / 4 + 255) / 256, 256, 0, stream>>>(x, xbf, nx / 4);
    transpose_convert_kernel<<<dim3(3 * DQ / 32, KQ / 32), 256, 0, stream>>>(w_qkv,  wqkvt,  KQ, 3 * DQ);
    transpose_convert_kernel<<<dim3(DQ / 32,     KQ / 32), 256, 0, stream>>>(w_proj, wprojt, KQ, DQ);

    // QKV GEMM: 4096x3072, tiles 128x128, TDM-staged
    wmma_gemm_kernel<<<dim3(3 * DQ / 128, MQ / 128), 256, 0, stream>>>(
        xbf, wqkvt, b_qkv, 3 * DQ, 0, nullptr, qbf, kbf, vbf, present);

    // flash attention: B*H heads x (S/64) query blocks, 4 waves of 16 rows each
    flash_attn_kernel<<<dim3(BQ * HQ * (SQ / 64)), 128, 0, stream>>>(qbf, kbf, vbf, attnbf);

    // projection GEMM: 4096x1024 -> fp32 out
    wmma_gemm_kernel<<<dim3(DQ / 128, MQ / 128), 256, 0, stream>>>(
        attnbf, wprojt, b_proj, DQ, 1, out, nullptr, nullptr, nullptr, nullptr);
}
